// VQ_21182778704209
// MI455X (gfx1250) — matrix operand order repaired and checked by reading, not compiled, not run on hardware
//
#include <hip/hip_runtime.h>
#include <stdint.h>

typedef __attribute__((ext_vector_type(16))) __bf16 v16bf;
typedef __attribute__((ext_vector_type(8)))  float  v8f;

#define NUM_E   1024
#define EDIM    64
#define CHW     262144     // C*H*W = 64*4096
#define HW      4096       // H*W
#define NTOT    65536      // B*H*W
#define ZNUMEL  4194304    // B*C*H*W

// ws layout (dwords): [0,32768) bpack, [32768,33792) enorm, then partials.
#define BPACK_DW   32768
#define ENORM_DW   1024
#define STAGE_B    (4 * (BPACK_DW + ENORM_DW))   // 135168 bytes staged to LDS

union BF16x16 { v16bf v; uint32_t u[8]; };

__device__ __forceinline__ uint32_t bf16rne(float f) {
  uint32_t u = __float_as_uint(f);
  return (u + 0x7FFFu + ((u >> 16) & 1u)) >> 16;
}
__device__ __forceinline__ uint32_t pk_bf16(float lo, float hi) {
  return bf16rne(lo) | (bf16rne(hi) << 16);
}

// ---------------------------------------------------------------------------
// Prep: pack codebook into WMMA B-fragment layout (bf16) + compute ||e||^2.
// B fragment (32x16 bf16, wave32): dword pair d holds K=2d,2d+1 for lanes
// 0-15 (K 0..15) and K=16+2d,16+2d+1 for lanes 16-31 (K 16..31); N = lane&15.
// LDS-bank-friendly layout: fragment f = j*2+t occupies 1024B as two 512B
// planes: byte = f*1024 + plane*512 + lane*16 (+dword*4), plane 0 = dwords
// 0..3, plane 1 = dwords 4..7. Lane stride 16B -> all 64 banks hit evenly.
// ---------------------------------------------------------------------------
__global__ void vq_prep(const float* __restrict__ emb,
                        uint32_t* __restrict__ bpack,
                        float* __restrict__ enorm) {
  int tid  = blockIdx.x * blockDim.x + threadIdx.x;   // 0..4095
  int j    = tid >> 6;
  int t    = (tid >> 5) & 1;
  int lane = tid & 31;
  int code  = 16 * j + (lane & 15);
  int cbase = t * 32 + ((lane >= 16) ? 16 : 0);
  const float* er = emb + (size_t)code * EDIM + cbase;
  const int frag = j * 2 + t;
  uint32_t* p0 = bpack + (size_t)frag * 256 + lane * 4;        // plane 0
  uint32_t* p1 = p0 + 128;                                     // plane 1
#pragma unroll
  for (int d = 0; d < 4; ++d) {
    p0[d] = pk_bf16(er[2 * d],     er[2 * d + 1]);
    p1[d] = pk_bf16(er[2 * d + 8], er[2 * d + 9]);
  }
  if (tid < NUM_E) {
    const float* r = emb + (size_t)tid * EDIM;
    float s = 0.f;
#pragma unroll
    for (int c = 0; c < EDIM; ++c) s = fmaf(r[c], r[c], s);
    enorm[tid] = s;
  }
}

// ---------------------------------------------------------------------------
// Main: per wave, 16 z-rows vs all 1024 codes via v_wmma_f32_16x16x32_bf16.
// Codebook fragments + ||e||^2 staged WG-wide into LDS with
// GLOBAL_LOAD_ASYNC_TO_LDS_B128 (ASYNCcnt path), then the GEMM loop reads
// B from DS while VMEM handles only z / z_q.
// ---------------------------------------------------------------------------
__global__ __launch_bounds__(256)
void vq_main(const float* __restrict__ z,
             const float* __restrict__ emb,
             const uint32_t* __restrict__ ws_stage,   // bpack + enorm, contiguous
             float* __restrict__ zq_out,
             int*   __restrict__ idx_out,
             float* __restrict__ partial) {
  __shared__ int sidx[128];
  extern __shared__ char smem[];   // STAGE_B bytes: bpack(128KB) + enorm(4KB)

  const int tid  = threadIdx.x;
  const int lane = tid & 31;
  const int wave = tid >> 5;

  // ---- Stage codebook into LDS: 33 x (512B per wave32 instruction) --------
  {
    const char* gsrc = (const char*)ws_stage;
    const uint32_t ldsbase = (uint32_t)(uintptr_t)smem;  // low 32b = LDS offset
#pragma unroll
    for (int i = 0; i < 33; ++i) {                        // 33*4096 = 135168 B
      const int off = tid * 16 + i * 4096;
      uint64_t ga = (uint64_t)(uintptr_t)(gsrc + off);
      uint32_t la = ldsbase + (uint32_t)off;
      asm volatile("global_load_async_to_lds_b128 %0, %1, off"
                   :: "v"(la), "v"(ga) : "memory");
    }
    asm volatile("s_wait_asynccnt 0x0" ::: "memory");
  }
  __syncthreads();

  const int4*  bls  = (const int4*)smem;                 // 64 int4 per fragment
  const float* enls = (const float*)(smem + 4 * BPACK_DW);

  const int n0  = blockIdx.x * 128 + wave * 16;   // 16 rows per wave
  const int b   = n0 >> 12;                       // 4096 rows per batch image
  const int rem = n0 & 4095;
  const int m   = lane & 15;                      // row-in-tile / col-in-tile
  const float* zb = z + (size_t)b * CHW + rem + m;

  // A fragment (16x32 bf16): lanes<16 rows M=lane hold K {0..15};
  // lanes 16-31 hold K {16..31}.
  BF16x16 afr[2];
  const int kb8 = (lane >= 16) ? 8 : 0;
#pragma unroll
  for (int t = 0; t < 2; ++t) {
    float f[16];
#pragma unroll
    for (int e = 0; e < 16; ++e) {
      int k = 2 * kb8 + e;
      int c = t * 32 + k;
      f[e] = zb[(size_t)c * HW];
    }
#pragma unroll
    for (int d = 0; d < 8; ++d) afr[t].u[d] = pk_bf16(f[2 * d], f[2 * d + 1]);
  }

  float minv[8];
  int   mini[8];
#pragma unroll
  for (int r = 0; r < 8; ++r) { minv[r] = 3.4e38f; mini[r] = 0; }

  for (int j = 0; j < 64; ++j) {                 // 64 tiles of 16 codes
    const int col = 16 * j + m;
    const float en = enls[col];
    v8f acc = {0.f, 0.f, 0.f, 0.f, 0.f, 0.f, 0.f, 0.f};
#pragma unroll
    for (int t = 0; t < 2; ++t) {
      BF16x16 bfr;
      const int fb = (j * 2 + t) * 64 + lane;    // int4 units
      int4 lo = bls[fb];
      int4 hi = bls[fb + 32];
      bfr.u[0] = lo.x; bfr.u[1] = lo.y; bfr.u[2] = lo.z; bfr.u[3] = lo.w;
      bfr.u[4] = hi.x; bfr.u[5] = hi.y; bfr.u[6] = hi.z; bfr.u[7] = hi.w;
      acc = __builtin_amdgcn_wmma_f32_16x16x32_bf16(
          false, afr[t].v, false, bfr.v, (short)0, acc, false, false);
    }
#pragma unroll
    for (int r = 0; r < 8; ++r) {
      float d = fmaf(acc[r], -2.0f, en);
      bool c = d < minv[r];
      minv[r] = c ? d   : minv[r];
      mini[r] = c ? col : mini[r];
    }
  }

  // Reduce argmin across each 16-lane half (rows 0-7 in lanes 0-15, 8-15 above)
#pragma unroll
  for (int r = 0; r < 8; ++r) {
#pragma unroll
    for (int msk = 1; msk <= 8; msk <<= 1) {
      float ov = __shfl_xor(minv[r], msk, 32);
      int   oi = __shfl_xor(mini[r], msk, 32);
      if (ov < minv[r] || (ov == minv[r] && oi < mini[r])) {
        minv[r] = ov; mini[r] = oi;
      }
    }
    if (lane == 0)  { sidx[wave * 16 + r]     = mini[r]; idx_out[n0 + r]     = mini[r]; }
    if (lane == 16) { sidx[wave * 16 + 8 + r] = mini[r]; idx_out[n0 + 8 + r] = mini[r]; }
  }
  __syncthreads();

  // Gather embedding rows -> z_q, accumulate squared error.
  const int myIdx = sidx[wave * 16 + m];
  const float* erow = emb + (size_t)myIdx * EDIM;
  float* qb = zq_out + (size_t)b * CHW + rem + m;
  const int chalf = (lane >= 16) ? 32 : 0;
  float lsum = 0.f;
  for (int cc = 0; cc < 32; ++cc) {
    int c = cc + chalf;
    size_t off = (size_t)c * HW;
    float e  = erow[c];
    float zv = zb[off];
    qb[off] = e;
    float d = e - zv;
    lsum = fmaf(d, d, lsum);
  }
#pragma unroll
  for (int msk = 16; msk >= 1; msk >>= 1) lsum += __shfl_xor(lsum, msk, 32);
  if (lane == 0) partial[blockIdx.x * 8 + wave] = lsum;
}

// ---------------------------------------------------------------------------
// Deterministic final loss reduction: loss = (1 + 0.25) * sum / numel
// ---------------------------------------------------------------------------
__global__ void vq_reduce(const float* __restrict__ partial,
                          float* __restrict__ loss) {
  __shared__ float s[256];
  float acc = 0.f;
  for (int i = threadIdx.x; i < 4096; i += 256) acc += partial[i];
  s[threadIdx.x] = acc;
  __syncthreads();
  for (int st = 128; st > 0; st >>= 1) {
    if ((int)threadIdx.x < st) s[threadIdx.x] += s[threadIdx.x + st];
    __syncthreads();
  }
  if (threadIdx.x == 0) loss[0] = 1.25f * s[0] / 4194304.0f;
}

extern "C" void kernel_launch(void* const* d_in, const int* in_sizes, int n_in,
                              void* d_out, int out_size, void* d_ws, size_t ws_size,
                              hipStream_t stream) {
  const float* z   = (const float*)d_in[0];
  const float* emb = (const float*)d_in[1];
  float* out = (float*)d_out;

  uint32_t* bpack  = (uint32_t*)d_ws;                       // 32768 dwords
  float* enorm     = (float*)d_ws + BPACK_DW;               // 1024 floats
  float* partial   = (float*)d_ws + BPACK_DW + ENORM_DW;    // 4096 floats

  float* zq   = out;
  int*   idxo = (int*)(out + ZNUMEL);
  float* loss = out + ZNUMEL + NTOT;

  (void)hipFuncSetAttribute((const void*)vq_main,
                            hipFuncAttributeMaxDynamicSharedMemorySize,
                            STAGE_B);

  vq_prep  <<<16,  256, 0, stream>>>(emb, bpack, enorm);
  vq_main  <<<512, 256, STAGE_B, stream>>>(z, emb, (const uint32_t*)d_ws,
                                           zq, idxo, partial);
  vq_reduce<<<1,   256, 0, stream>>>(partial, loss);
}
